// SelfAttentionBlock_66468913873239
// MI455X (gfx1250) — compile-verified, matrix operand order checked by reading, not run on hardware
//
#include <hip/hip_runtime.h>
#include <hip/hip_fp16.h>
#include <stdint.h>

typedef __attribute__((ext_vector_type(16))) _Float16 v16h;
typedef __attribute__((ext_vector_type(8)))  float    v8f;

#define N_TOK 8192
#define C_CH  32
#define HD    8
#define NH    4

union F16Frag {
    v16h      v;
    _Float16  h[16];
    uint4     q[2];   // two 16-byte halves (elements 0-7, 8-15)
};

__device__ __forceinline__ float xor16_f32(float x) {
    // lane <-> lane^16 swap: group-of-32 swizzle, xor=0x10, and=0x1f
    return __int_as_float(__builtin_amdgcn_ds_swizzle(__float_as_int(x), 0x401F));
}

// ---------------------------------------------------------------------------
// Kernel 1: QKV projection (1x1x1 conv == 32x32 channel matmul), fp32 in,
// f16 out in WMMA-friendly layouts:
//   Qh[h][n][d]  (pre-scaled by 1/sqrt(8))
//   Kh[h][n][d]
//   Vt[h][d][n]  (transposed so the V^T WMMA A-operand loads are contiguous)
// ---------------------------------------------------------------------------
__global__ __launch_bounds__(256) void qkv_proj_kernel(
    const float* __restrict__ qf, const float* __restrict__ kf,
    const float* __restrict__ Wq, const float* __restrict__ Wk,
    const float* __restrict__ Wv,
    _Float16* __restrict__ Qh, _Float16* __restrict__ Kh,
    _Float16* __restrict__ Vt)
{
    const int n = blockIdx.x * blockDim.x + threadIdx.x;
    float qin[C_CH], kin[C_CH];
#pragma unroll
    for (int c = 0; c < C_CH; ++c) {
        qin[c] = qf[c * N_TOK + n];
        kin[c] = kf[c * N_TOK + n];
    }
    const float qscale = 0.35355339059327373f;  // 8^-0.5
#pragma unroll 4
    for (int o = 0; o < C_CH; ++o) {
        float aq = 0.f, ak = 0.f, av = 0.f;
#pragma unroll
        for (int c = 0; c < C_CH; ++c) {
            aq = fmaf(Wq[o * C_CH + c], qin[c], aq);
            ak = fmaf(Wk[o * C_CH + c], kin[c], ak);
            av = fmaf(Wv[o * C_CH + c], kin[c], av);
        }
        const int h = o >> 3, d = o & 7;
        Qh[(h * N_TOK + n) * HD + d] = (_Float16)(aq * qscale);
        Kh[(h * N_TOK + n) * HD + d] = (_Float16)ak;
        Vt[(h * HD + d) * N_TOK + n] = (_Float16)av;
    }
}

// ---------------------------------------------------------------------------
// Kernel 2: flash attention, one wave32 per (head, 16-query tile).
//   S^T(16 keys x 16 queries) = KeyTile(16x32, K=hd padded) x Q^T(32x16)
//   ctx^T(hd x 16 queries)   += V^T(16x32, K=keys)          x P^T(32x16)
//
// The Q B-operand has all K=8..31 matrix rows zero (zeros live in registers
// set once before the loop), so the K-tile A-operand may carry arbitrary
// *finite* data in its K=8..31 columns -> lanes 16-31 load duplicate valid
// K rows unconditionally (no EXEC predication in the loop). Fragment elements
// 8-15 are zeroed once outside the loop and never overwritten (NaN-garbage
// would poison WMMA since NaN*0=NaN). V^T rows 8-15 are clamped duplicates;
// their D rows are never stored.
// ---------------------------------------------------------------------------
__global__ __launch_bounds__(256) void attn_kernel(
    const _Float16* __restrict__ Qh, const _Float16* __restrict__ Kh,
    const _Float16* __restrict__ Vt, float* __restrict__ ctxbuf)
{
    const int tid   = blockIdx.x * blockDim.x + threadIdx.x;
    const int wave  = tid >> 5;
    const int lane  = tid & 31;
    const int head  = wave >> 9;          // 512 query tiles per head
    const int qtile = wave & 511;
    const int qbase = qtile * 16;
    const int lq    = lane & 15;
    const int half  = lane >> 4;

    // Q^T as WMMA B operand: lane L = query lq; elems 0-7 = hd 0-7 (lanes<16),
    // everything else (hd pad rows 8-31) zero.
    F16Frag bq;
#pragma unroll
    for (int i = 0; i < 16; ++i) bq.h[i] = (_Float16)0.0f;
    if (lane < 16)
        bq.q[0] = *reinterpret_cast<const uint4*>(
            Qh + (size_t)(head * N_TOK + qbase + lq) * HD);

    // K-tile fragments: zero elements 8-15 once; loop loads touch only q[0].
    F16Frag a0, a1, va;
#pragma unroll
    for (int i = 0; i < 16; ++i) { a0.h[i] = (_Float16)0.0f; a1.h[i] = (_Float16)0.0f; }

    v8f   ctx = {};
    float m = -__builtin_inff();
    float l = 0.0f;

    const _Float16* kptr = Kh + (size_t)head * N_TOK * HD;
    const _Float16* vptr = Vt + (size_t)(head * HD + (lq & 7)) * N_TOK; // rows 8-15 clamped

    for (int kt = 0; kt < N_TOK / 32; ++kt) {
        const int kb = kt * 32;

        // Key tiles as A operands: all 32 lanes load row (kb [+16] + lq);
        // lanes 16-31 carry duplicates nullified by bq's zero K rows.
        a0.q[0] = *reinterpret_cast<const uint4*>(kptr + (size_t)(kb + lq) * HD);
        a1.q[0] = *reinterpret_cast<const uint4*>(kptr + (size_t)(kb + 16 + lq) * HD);

        v8f z = {};
        v8f st0 = __builtin_amdgcn_wmma_f32_16x16x32_f16(false, a0.v, false, bq.v,
                                                         (short)0, z, false, false);
        v8f st1 = __builtin_amdgcn_wmma_f32_16x16x32_f16(false, a1.v, false, bq.v,
                                                         (short)0, z, false, false);

        // st*[r] = S^T[key = kb + r + 8*half (+16 for st1)][query = lq]
        // Balanced-tree max over the 16 per-lane key scores (short dep chain).
        float mx[8];
#pragma unroll
        for (int r = 0; r < 8; ++r) mx[r] = fmaxf(st0[r], st1[r]);
#pragma unroll
        for (int r = 0; r < 4; ++r) mx[r] = fmaxf(mx[r], mx[r + 4]);
        float tmax = fmaxf(fmaxf(mx[0], mx[1]), fmaxf(mx[2], mx[3]));
        tmax = fmaxf(tmax, xor16_f32(tmax));            // other half's keys
        const float mnew  = fmaxf(m, tmax);
        const float alpha = __expf(m - mnew);           // exp(-inf)=0 on 1st iter
        m = mnew;

        // P^T values drop straight into the B-operand layout of the PV WMMA.
        F16Frag pb;
        float s[8];
#pragma unroll
        for (int r = 0; r < 8; ++r) {
            const float p0 = __expf(st0[r] - mnew);
            const float p1 = __expf(st1[r] - mnew);
            s[r] = p0 + p1;
            pb.h[r]     = (_Float16)p0;
            pb.h[r + 8] = (_Float16)p1;
        }
#pragma unroll
        for (int r = 0; r < 4; ++r) s[r] += s[r + 4];
        float psum = (s[0] + s[1]) + (s[2] + s[3]);
        psum += xor16_f32(psum);
        l = l * alpha + psum;

#pragma unroll
        for (int r = 0; r < 8; ++r) ctx[r] *= alpha;

        // V^T tile as A operand: M = hd, K = 32 keys; both halves loaded.
        va.q[0] = *reinterpret_cast<const uint4*>(vptr + kb + 8 * half);
        va.q[1] = *reinterpret_cast<const uint4*>(vptr + kb + 16 + 8 * half);

        ctx = __builtin_amdgcn_wmma_f32_16x16x32_f16(false, va.v, false, pb.v,
                                                     (short)0, ctx, false, false);

        if (kt + 1 < N_TOK / 32) {
            // locality 3 -> near (WGP) scope: pull next tiles toward L0 under
            // the softmax VALU shadow (stream is L2-resident anyway).
            __builtin_prefetch(kptr + (size_t)(kb + 32 + lq) * HD, 0, 3);
            __builtin_prefetch(vptr + kb + 32 + 8 * half, 0, 3);
        }
    }

    // ctx^T: VGPR d, lanes 0-15 -> (hd=d, query=lq). Normalize and store as
    // ctxbuf[C][N] with C = head*8 + d  (matches reference reshape order).
    if (lane < 16) {
        const float invl = 1.0f / l;
#pragma unroll
        for (int d = 0; d < HD; ++d)
            ctxbuf[(size_t)(head * HD + d) * N_TOK + qbase + lq] = ctx[d] * invl;
    }
}

// ---------------------------------------------------------------------------
// Kernel 3: output projection out = Wo (32x32) x ctx (32xN)
// ---------------------------------------------------------------------------
__global__ __launch_bounds__(256) void out_proj_kernel(
    const float* __restrict__ ctxbuf, const float* __restrict__ Wo,
    float* __restrict__ out)
{
    const int n = blockIdx.x * blockDim.x + threadIdx.x;
    float cin[C_CH];
#pragma unroll
    for (int c = 0; c < C_CH; ++c) cin[c] = ctxbuf[c * N_TOK + n];
#pragma unroll 4
    for (int o = 0; o < C_CH; ++o) {
        float acc = 0.f;
#pragma unroll
        for (int c = 0; c < C_CH; ++c)
            acc = fmaf(Wo[o * C_CH + c], cin[c], acc);
        out[o * N_TOK + n] = acc;
    }
}

extern "C" void kernel_launch(void* const* d_in, const int* in_sizes, int n_in,
                              void* d_out, int out_size, void* d_ws, size_t ws_size,
                              hipStream_t stream) {
    const float* qf = (const float*)d_in[0];
    const float* kf = (const float*)d_in[1];
    const float* Wq = (const float*)d_in[2];
    const float* Wk = (const float*)d_in[3];
    const float* Wv = (const float*)d_in[4];
    const float* Wo = (const float*)d_in[5];
    float* out = (float*)d_out;

    char* ws = (char*)d_ws;
    _Float16* Qh = (_Float16*)(ws);                    // 4*8192*8*2 = 512 KB
    _Float16* Kh = (_Float16*)(ws +  512 * 1024);      // 512 KB
    _Float16* Vt = (_Float16*)(ws + 1024 * 1024);      // 512 KB
    float*    cx = (float*)   (ws + 1536 * 1024);      // 32*8192*4 = 1 MB

    qkv_proj_kernel<<<N_TOK / 256, 256, 0, stream>>>(qf, kf, Wq, Wk, Wv, Qh, Kh, Vt);
    // 4 heads * 512 query tiles = 2048 waves = 65536 threads
    attn_kernel<<<(NH * (N_TOK / 16) * 32) / 256, 256, 0, stream>>>(Qh, Kh, Vt, cx);
    out_proj_kernel<<<N_TOK / 256, 256, 0, stream>>>(cx, Wo, out);
}
